// CRF_68118181314925
// MI455X (gfx1250) — matrix operand order repaired and checked by reading, not compiled, not run on hardware
//
#include <hip/hip_runtime.h>
#include <hip/hip_bf16.h>
#include <math.h>

typedef __attribute__((ext_vector_type(16))) _Float16 v16h;
typedef __attribute__((ext_vector_type(8)))  _Float16 v8h;
typedef __attribute__((ext_vector_type(8)))  float    v8f;

#define TT 1024
#define BB 128
#define LL 128
#define ROWS 16   // batch rows per workgroup (one 16-row WMMA tile)

#define LOG2E 1.4426950408889634f
#define LN2   0.6931471805599453f

__device__ __forceinline__ float fast_exp(float x) {
    return __builtin_amdgcn_exp2f(x * LOG2E);     // bare v_exp_f32
}
__device__ __forceinline__ float fast_log(float x) {
    return __builtin_amdgcn_logf(x) * LN2;        // bare v_log_f32
}

// ---------------------------------------------------------------------------
// Denominator: CRF forward recursion in exp-space via WMMA f16 GEMMs.
// Grid: BB/ROWS = 8 blocks of 256 threads (8 waves). Wave w owns column tile
// [16w, 16w+16). B-fragments (exp'd transitions) live in registers for all T.
// Two barriers per timestep.
// ---------------------------------------------------------------------------
__global__ __launch_bounds__(256)
void crf_forward_kernel(const float* __restrict__ pred,
                        const int*   __restrict__ mask,
                        const float* __restrict__ trans,
                        const float* __restrict__ start,
                        const float* __restrict__ endv,
                        float*       __restrict__ den)
{
    __shared__ float    state[ROWS][LL];   // running scores (f32)      8 KB
    __shared__ _Float16 Ash[ROWS][LL];     // exp'd A tile (f16)        4 KB
    __shared__ float    tmax[LL];          // per-column max of trans
    __shared__ float    rowmax[ROWS];      // per-row max of state
    __shared__ int      mrow[ROWS];        // mask for current step

    const int tid  = threadIdx.x;
    const int wave = tid >> 5;             // 0..7 -> column tile index
    const int lane = tid & 31;
    const int b0   = blockIdx.x * ROWS;

    // ---- per-column max of transitions ------------------------------------
    if (tid < LL) {
        float mx = -INFINITY;
        for (int i = 0; i < LL; ++i) mx = fmaxf(mx, trans[i * LL + tid]);
        tmax[tid] = mx;
    }
    __syncthreads();

    // ---- build persistent B fragments: P[i,j] = exp(trans[i,j]-tmax[j]) ---
    const int nloc  = lane & 15;
    const int khalf = lane >> 4;
    const int Ncol  = wave * 16 + nloc;
    const float tmaxN = tmax[Ncol];

    v16h Bfrag[4];
    #pragma unroll
    for (int kt = 0; kt < 4; ++kt) {
        const int Kbase = kt * 32 + khalf * 16;     // 16 consecutive K per lane
        #pragma unroll
        for (int k2 = 0; k2 < 16; ++k2) {
            float p = fast_exp(trans[(Kbase + k2) * LL + Ncol] - tmaxN);
            Bfrag[kt][k2] = (_Float16)p;
        }
    }

    // ---- init: score0[b,j] = start[j] + pred[0,b,j] ------------------------
    for (int e = tid; e < ROWS * LL; e += 256) {
        const int r = e >> 7, j = e & 127;
        state[r][j] = start[j] + pred[(size_t)(b0 + r) * LL + j];
    }
    __syncthreads();

    // ---- forward recursion over time --------------------------------------
    for (int t = 1; t < TT; ++t) {
        // stage 1+2 fused: row max (butterfly leaves mx in all 16 lanes),
        // then A = exp(state - mx) packed to f16 and stored as one b128.
        {
            const int r = tid >> 4, c0 = (tid & 15) * 8;
            float vals[8];
            #pragma unroll
            for (int c = 0; c < 8; ++c) vals[c] = state[r][c0 + c];
            float mx = vals[0];
            #pragma unroll
            for (int c = 1; c < 8; ++c) mx = fmaxf(mx, vals[c]);
            #pragma unroll
            for (int off = 8; off >= 1; off >>= 1)
                mx = fmaxf(mx, __shfl_xor(mx, off, 16));
            if ((tid & 15) == 0) rowmax[r] = mx;
            v8h a;
            #pragma unroll
            for (int c = 0; c < 8; ++c)
                a[c] = (_Float16)fast_exp(vals[c] - mx);
            *(v8h*)&Ash[r][c0] = a;
        }
        if (tid < ROWS) mrow[tid] = mask[t * BB + b0 + tid];
        __syncthreads();

        // stage 3: (16x128)x(128x16) GEMM via 4 WMMAs, f32 accumulate
        v8f acc = {};
        {
            const int m  = lane & 15;
            const int hi = lane >> 4;
            #pragma unroll
            for (int kt = 0; kt < 4; ++kt) {
                const int K0 = kt * 32 + hi * 16;
                v8h a0 = *(const v8h*)&Ash[m][K0];        // K0..K0+7
                v8h a1 = *(const v8h*)&Ash[m][K0 + 8];    // K0+8..K0+15
                v16h a = __builtin_shufflevector(a0, a1,
                        0,1,2,3,4,5,6,7,8,9,10,11,12,13,14,15);
                acc = __builtin_amdgcn_wmma_f32_16x16x32_f16(
                        false, a, false, Bfrag[kt], (short)0, acc, false, false);
            }
        }

        // stage 4: score' = rowmax[b] + tmax[j] + log(acc) + pred[t,b,j]
        {
            const int hi = lane >> 4;
            const float* pbase =
                pred + ((size_t)t * BB + b0 + 8 * hi) * LL + wave * 16 + nloc;
            #pragma unroll
            for (int v = 0; v < 8; ++v) {
                const int M = v + 8 * hi;
                const float oldv = state[M][wave * 16 + nloc];
                const float nv =
                    rowmax[M] + tmaxN + fast_log(acc[v]) + pbase[(size_t)v * LL];
                state[M][wave * 16 + nloc] = mrow[M] ? nv : oldv;
            }
        }
        __syncthreads();
    }

    // ---- final: den[b] = logsumexp_j(state[b,j] + end[j]) ------------------
    {
        const int r = tid >> 4, c0 = (tid & 15) * 8;
        float vals[8];
        float mx = -INFINITY;
        #pragma unroll
        for (int c = 0; c < 8; ++c) {
            vals[c] = state[r][c0 + c] + endv[c0 + c];
            mx = fmaxf(mx, vals[c]);
        }
        #pragma unroll
        for (int off = 8; off >= 1; off >>= 1)
            mx = fmaxf(mx, __shfl_xor(mx, off, 16));
        float s = 0.f;
        #pragma unroll
        for (int c = 0; c < 8; ++c) s += fast_exp(vals[c] - mx);
        #pragma unroll
        for (int off = 8; off >= 1; off >>= 1) s += __shfl_xor(s, off, 16);
        if ((tid & 15) == 0) den[b0 + r] = mx + fast_log(s);
    }
}

// ---------------------------------------------------------------------------
// Numerator: one wave per batch element (128 waves = 16 blocks x 8 waves);
// lanes stride over t, butterfly reduce score and mask-sum.
// ---------------------------------------------------------------------------
__global__ __launch_bounds__(256)
void crf_num_kernel(const float*     __restrict__ pred,
                    const long long* __restrict__ targets,
                    const int*       __restrict__ mask,
                    const float*     __restrict__ trans,
                    const float*     __restrict__ start,
                    const float*     __restrict__ endv,
                    float*           __restrict__ num)
{
    const int b    = blockIdx.x * 8 + (threadIdx.x >> 5);  // 0..127
    const int lane = threadIdx.x & 31;

    float s   = 0.f;
    int msum  = 0;
    for (int t = 1 + lane; t < TT; t += 32) {
        const long long tp = targets[(size_t)(t - 1) * BB + b];
        const long long tg = targets[(size_t)t * BB + b];
        const int mk = mask[t * BB + b];
        msum += mk;
        const float term = trans[tp * LL + tg] +
                           pred[((size_t)t * BB + b) * LL + tg];
        s += mk ? term : 0.f;
    }
    #pragma unroll
    for (int off = 16; off >= 1; off >>= 1) {
        s    += __shfl_xor(s, off, 32);
        msum += __shfl_xor(msum, off, 32);
    }
    if (lane == 0) {
        msum += mask[b];                                   // t = 0 term
        const long long t0 = targets[b];
        s += start[t0] + pred[(size_t)b * LL + t0];
        const int last_idx = msum - 1;
        s += endv[targets[(size_t)last_idx * BB + b]];
        num[b] = s;
    }
}

// ---------------------------------------------------------------------------
// Final: out = mean(den - num) over B.
// ---------------------------------------------------------------------------
__global__ __launch_bounds__(128)
void crf_final_kernel(const float* __restrict__ den,
                      const float* __restrict__ num,
                      float*       __restrict__ out)
{
    const int b = threadIdx.x;
    float diff = den[b] - num[b];
    __shared__ float red[4];
    #pragma unroll
    for (int off = 16; off >= 1; off >>= 1) diff += __shfl_xor(diff, off, 32);
    if ((b & 31) == 0) red[b >> 5] = diff;
    __syncthreads();
    if (b == 0)
        out[0] = (red[0] + red[1] + red[2] + red[3]) / (float)BB;
}

// ---------------------------------------------------------------------------
extern "C" void kernel_launch(void* const* d_in, const int* in_sizes, int n_in,
                              void* d_out, int out_size, void* d_ws, size_t ws_size,
                              hipStream_t stream)
{
    const float*     pred    = (const float*)d_in[0];      // (T,B,L) f32
    const long long* targets = (const long long*)d_in[1];  // (T,B)  i64
    const int*       mask    = (const int*)d_in[2];        // (T,B)  i32
    const float*     trans   = (const float*)d_in[3];      // (L,L)  f32
    const float*     start   = (const float*)d_in[4];      // (L,)   f32
    const float*     endv    = (const float*)d_in[5];      // (L,)   f32

    float* den = (float*)d_ws;                 // ws[0..127]
    float* num = ((float*)d_ws) + BB;          // ws[128..255]
    float* out = (float*)d_out;                // single float

    crf_num_kernel<<<BB / 8, 256, 0, stream>>>(pred, targets, mask, trans,
                                               start, endv, num);
    crf_forward_kernel<<<BB / ROWS, 256, 0, stream>>>(pred, mask, trans,
                                                      start, endv, den);
    crf_final_kernel<<<1, 128, 0, stream>>>(den, num, out);
}
